// localUp_48017734369836
// MI455X (gfx1250) — matrix operand order repaired and checked by reading, not compiled, not run on hardware
//
#include <hip/hip_runtime.h>
#include <hip/hip_bf16.h>
#include <math.h>

typedef __attribute__((ext_vector_type(16))) _Float16 v16h;
typedef __attribute__((ext_vector_type(8)))  float    v8f;
typedef __attribute__((ext_vector_type(4)))  unsigned int u32x4;
typedef __attribute__((ext_vector_type(8)))  int      i32x8;
typedef __attribute__((ext_vector_type(4)))  int      i32x4;

#define NIMG 4
#define HOUT 128
#define WOUT 128
#define HWOUT (HOUT*WOUT)      // 16384
#define PTOT (NIMG*HWOUT)      // 65536 output pixels
#define HS 64
#define WS 64
#define HWS (HS*WS)            // 4096
#define P2TOT (NIMG*HWS)       // 16384 source pixels
#define CK 32                  // refine channels
#define COUT 256

#if __has_builtin(__builtin_amdgcn_tensor_load_to_lds) && \
    __has_builtin(__builtin_amdgcn_s_wait_tensorcnt)
#define ATHENA_TDM 1
#else
#define ATHENA_TDM 0
#endif

// ---------------------------------------------------------------------------
// WMMA GEMM (+ optional per-pixel L2 normalize over CK=32 output channels).
//   y[p][k] = sum_c x[n][c][pix(p)] * w[k][c]
// x: [NIMG][C][HWPIX] planar fp32, w: [CK][C] fp32, y: pixel-major [P][CK] fp32
//
// Data movement: the Tensor Data Mover DMAs a [KC=64 ch][128 px] fp32 tile
// (32 KB) from global to LDS per 2-D descriptor, double-buffered and tracked
// with TENSORcnt (one chunk in flight while WMMA consumes the other).
// One wave = 16 pixels x 32 out-channels (two v_wmma_f32_16x16x32_f16/K-step).
// ---------------------------------------------------------------------------
template<int C, int HWPIX, bool NORM>
__global__ __launch_bounds__(256)
void wmma_refine_kernel(const float* __restrict__ x,
                        const float* __restrict__ w,
                        float* __restrict__ y)
{
    constexpr int KC  = 64;        // channels per DMA chunk
    constexpr int NCH = C / KC;    // chunks per block
    __shared__ __attribute__((aligned(128))) float    abuf[2][KC * 128];
    __shared__ __attribute__((aligned(32)))  _Float16 wlds[CK * C];

    const int tid  = threadIdx.x;
    const int lane = tid & 31;
    const int wave = tid >> 5;
    const bool w0  = (wave == 0);          // wave-uniform

    const int p0blk   = blockIdx.x * 128;  // 128 pixels per block
    const int n       = p0blk / HWPIX;     // tiles never straddle images
    const int pixbase = p0blk % HWPIX;
    const float* xi   = x + (size_t)n * C * HWPIX + pixbase;

#if ATHENA_TDM
    // Issue one TDM 2-D tile load: tile_dim0=128 contiguous pixels,
    // tile_dim1=KC rows strided by HWPIX, data_size=4B  (ISA 08_async_tensor §8)
    auto tdm_issue = [&](int ch) {
        unsigned long long ga = (unsigned long long)(uintptr_t)
                                (xi + (size_t)ch * KC * HWPIX);
        unsigned int ldsa = (unsigned int)(uintptr_t)&abuf[ch & 1][0];
        u32x4 g0 = { 1u,                                   // count=1, user desc
                     ldsa,                                 // lds_addr
                     (unsigned int)ga,                     // global_addr[31:0]
                     (unsigned int)(ga >> 32) | 0x80000000u }; // [56:32] | type=2
        i32x8 g1 = { (int)0x00020000,          // data_size=2 (4B), no pad/iter
                     (int)(128u << 16),        // tensor_dim0[15:0]=128
                     (int)((unsigned)KC << 16),// tensor_dim0 hi=0, dim1 lo=KC
                     (int)(128u << 16),        // dim1 hi=0, tile_dim0=128
                     (int)KC,                  // tile_dim1=KC, tile_dim2=0
                     (int)HWPIX,               // tensor_dim0_stride (elements)
                     0, 0 };
        i32x4 gz = { 0, 0, 0, 0 };
#if defined(__clang_major__) && __clang_major__ >= 23
        i32x8 gz8 = { 0, 0, 0, 0, 0, 0, 0, 0 };
        __builtin_amdgcn_tensor_load_to_lds(g0, g1, gz, gz, gz8, 0);
#else
        __builtin_amdgcn_tensor_load_to_lds(g0, g1, gz, gz, 0);
#endif
    };
    if (w0) tdm_issue(0);
#endif

    // stage weights into LDS as f16 (coalesced, converted once per block);
    // overlaps with the first TDM transfer
    for (int i = tid; i < CK * C; i += 256)
        wlds[i] = (_Float16)w[i];
    __syncthreads();

    // A-matrix 16x32 f16 layout (matches B's contiguous K order):
    //   lanes 0-15 : M=lane,    V0-7 = K 0..15
    //   lanes16-31 : M=lane-16, V0-7 = K 16..31
    const int mrow = lane & 15;
    const int klo  = (lane & 16) ? 16 : 0;
    // B-matrix 32x16 f16: lane = N column; lanes 0-15 hold K 0..15,
    // lanes 16-31 hold K 16..31, contiguous in V0-7.
    const int bK   = (lane & 16) ? 16 : 0;
    const int ncol = lane & 15;
    const int pl   = wave * 16 + mrow;     // local pixel within block tile

    v8f acc0 = {}; v8f acc1 = {};
    for (int ch = 0; ch < NCH; ++ch) {
#if ATHENA_TDM
        if (w0) {
            if (ch + 1 < NCH) { tdm_issue(ch + 1);
                                __builtin_amdgcn_s_wait_tensorcnt(1); }
            else              { __builtin_amdgcn_s_wait_tensorcnt(0); }
        }
#else
        {   // fallback: cooperative coalesced copy of the chunk
            float4* dst = (float4*)&abuf[ch & 1][0];
            for (int i = tid; i < KC * 128 / 4; i += 256) {
                int row = i >> 5, col = i & 31;
                dst[i] = *(const float4*)(xi + (size_t)(ch * KC + row) * HWPIX
                                             + col * 4);
            }
        }
#endif
        __syncthreads();                    // chunk ch visible to all waves

        const float* ab = &abuf[ch & 1][0];
        #pragma unroll
        for (int s = 0; s < KC / 32; ++s) {
            const int s32 = s * 32;
            v16h a;
            #pragma unroll
            for (int e = 0; e < 8; ++e) {   // conflict-free ds reads
                a[e]     = (_Float16)ab[(s32 + klo + e)      * 128 + pl];
                a[e + 8] = (_Float16)ab[(s32 + 8 + klo + e) * 128 + pl];
            }
            const int kbg = ch * KC + s32;
            v16h b0 = *(const v16h*)&wlds[(size_t)(ncol     ) * C + kbg + bK];
            v16h b1 = *(const v16h*)&wlds[(size_t)(ncol + 16) * C + kbg + bK];
            acc0 = __builtin_amdgcn_wmma_f32_16x16x32_f16(false, a, false, b0,
                                                          (short)0, acc0, false, false);
            acc1 = __builtin_amdgcn_wmma_f32_16x16x32_f16(false, a, false, b1,
                                                          (short)0, acc1, false, false);
        }
        __syncthreads();                    // all reads done before buffer reuse
    }

    // C/D layout: lane = N column (out-channel), VGPR r = M row.
    // lanes 0-15 hold pixel m=r, lanes 16-31 hold pixel m=r+8 -> xor-butterfly
    // over masks 1,2,4,8 reduces exactly one pixel's 32 channels (wave32).
    const int p0 = p0blk + wave * 16;
    #pragma unroll
    for (int r = 0; r < 8; ++r) {
        const int m = r + ((lane & 16) ? 8 : 0);
        float scale = 1.0f;
        if (NORM) {
            float s = acc0[r] * acc0[r] + acc1[r] * acc1[r];
            s += __shfl_xor(s, 1);
            s += __shfl_xor(s, 2);
            s += __shfl_xor(s, 4);
            s += __shfl_xor(s, 8);
            scale = 1.0f / fmaxf(sqrtf(s), 1e-12f);
        }
        float* yp = y + (size_t)(p0 + m) * CK;
        yp[ncol]      = acc0[r] * scale;
        yp[ncol + 16] = acc1[r] * scale;
    }
}

// ---------------------------------------------------------------------------
// Bilinear upsample (align_corners) 64->128 of the 32-channel key field,
// then L2-normalize per pixel.  kfs pixel-major [P2][32] -> kfn [P][32].
// (1x1 conv and bilinear upsample commute, so normalizing here matches ref.)
// ---------------------------------------------------------------------------
__global__ __launch_bounds__(256)
void upsample_norm_kernel(const float* __restrict__ kfs, float* __restrict__ kfn)
{
    const int p  = blockIdx.x * 256 + threadIdx.x;
    const int n  = p / HWOUT;
    const int rem = p % HWOUT;
    const int yy = rem / WOUT, xx = rem % WOUT;
    const float sc = 63.0f / 127.0f;
    const float fy = yy * sc, fx = xx * sc;
    const int y0 = (int)fy, x0 = (int)fx;
    const int y1 = min(y0 + 1, HS - 1), x1 = min(x0 + 1, WS - 1);
    const float wy = fy - (float)y0, wx = fx - (float)x0;
    const float w00 = (1.f - wy) * (1.f - wx), w01 = (1.f - wy) * wx;
    const float w10 = wy * (1.f - wx),         w11 = wy * wx;

    const float* base = kfs + (size_t)n * HWS * CK;
    const float* s00 = base + (size_t)(y0 * WS + x0) * CK;
    const float* s01 = base + (size_t)(y0 * WS + x1) * CK;
    const float* s10 = base + (size_t)(y1 * WS + x0) * CK;
    const float* s11 = base + (size_t)(y1 * WS + x1) * CK;

    float v[CK]; float ss = 0.f;
    #pragma unroll
    for (int c = 0; c < CK; ++c) {
        float t = w00 * s00[c] + w01 * s01[c] + w10 * s10[c] + w11 * s11[c];
        v[c] = t; ss += t * t;
    }
    const float scale = 1.0f / fmaxf(sqrtf(ss), 1e-12f);
    float* o = kfn + (size_t)p * CK;
    #pragma unroll
    for (int c = 0; c < CK; ++c) o[c] = v[c] * scale;
}

// ---------------------------------------------------------------------------
// energy[k] = q . kfn(neighbor k), dilation-2 3x3 with zero padding,
// then softmax over the 9 taps.  att pixel-major [P][9].
// ---------------------------------------------------------------------------
__global__ __launch_bounds__(256)
void energy_softmax_kernel(const float* __restrict__ qn,
                           const float* __restrict__ kfn,
                           float* __restrict__ att)
{
    const int p  = blockIdx.x * 256 + threadIdx.x;
    const int n  = p / HWOUT;
    const int rem = p % HWOUT;
    const int yy = rem / WOUT, xx = rem % WOUT;

    float q[CK];
    const float* qp = qn + (size_t)p * CK;
    #pragma unroll
    for (int c = 0; c < CK; ++c) q[c] = qp[c];

    float e[9];
    #pragma unroll
    for (int k = 0; k < 9; ++k) {
        const int dy = (k / 3) * 2 - 2, dx = (k % 3) * 2 - 2;
        const int ny = yy + dy, nx = xx + dx;
        float d = 0.f;
        if (ny >= 0 && ny < HOUT && nx >= 0 && nx < WOUT) {
            const float* kp = kfn + (size_t)(n * HWOUT + ny * WOUT + nx) * CK;
            #pragma unroll
            for (int c = 0; c < CK; ++c) d = fmaf(q[c], kp[c], d);
        }
        e[k] = d;   // zero-padded patches give energy 0 (matches unfold pad=2)
    }
    float m = e[0];
    #pragma unroll
    for (int k = 1; k < 9; ++k) m = fmaxf(m, e[k]);
    float s = 0.f;
    #pragma unroll
    for (int k = 0; k < 9; ++k) { e[k] = __expf(e[k] - m); s += e[k]; }
    const float inv = 1.0f / s;
    float* ap = att + (size_t)p * 9;
    #pragma unroll
    for (int k = 0; k < 9; ++k) ap[k] = e[k] * inv;
}

// ---------------------------------------------------------------------------
// Fused aggregation: result[n,c,y,x] = sum_k att[k] * bilinear(out)[c, y+dy, x+dx].
// The 67 MB upsampled value tensor is never materialized: each of the 9 taps is
// expanded into its 4 bilinear source taps (36 combined weights, channel-
// independent -> computed once per thread and reused across 128 channels).
// Block = one output row (n,y); lanes sweep x for coalescing.
// ---------------------------------------------------------------------------
__global__ __launch_bounds__(256)
void aggregate_kernel(const float* __restrict__ att,
                      const float* __restrict__ outsrc,
                      float* __restrict__ res)
{
    const int b  = blockIdx.x;          // n*HOUT + y
    const int n  = b / HOUT, yy = b % HOUT;
    const int xx = threadIdx.x & 127;
    const int cs = threadIdx.x >> 7;    // channel slot 0..1
    const int p  = n * HWOUT + yy * WOUT + xx;

    float a9[9];
    const float* ap = att + (size_t)p * 9;
    #pragma unroll
    for (int k = 0; k < 9; ++k) a9[k] = ap[k];

    int   toff[36];
    float twgt[36];
    const float sc = 63.0f / 127.0f;
    #pragma unroll
    for (int k = 0; k < 9; ++k) {
        const int dy = (k / 3) * 2 - 2, dx = (k % 3) * 2 - 2;
        const int ny = yy + dy, nx = xx + dx;
        const bool ok = (ny >= 0 && ny < HOUT && nx >= 0 && nx < WOUT);
        const float fy = ok ? ny * sc : 0.f, fx = ok ? nx * sc : 0.f;
        const int y0 = (int)fy, x0 = (int)fx;
        const int y1 = min(y0 + 1, HS - 1), x1 = min(x0 + 1, WS - 1);
        const float wy = fy - (float)y0, wx = fx - (float)x0;
        const float aw = ok ? a9[k] : 0.f;   // zero-padded neighbors contribute 0
        toff[k*4+0] = y0 * WS + x0;  twgt[k*4+0] = aw * (1.f - wy) * (1.f - wx);
        toff[k*4+1] = y0 * WS + x1;  twgt[k*4+1] = aw * (1.f - wy) * wx;
        toff[k*4+2] = y1 * WS + x0;  twgt[k*4+2] = aw * wy * (1.f - wx);
        toff[k*4+3] = y1 * WS + x1;  twgt[k*4+3] = aw * wy * wx;
    }

    const float* src = outsrc + (size_t)n * COUT * HWS;
    float* dst = res + (size_t)n * COUT * HWOUT + (size_t)yy * WOUT + xx;
    for (int i = 0; i < COUT / 2; ++i) {
        const int c = cs * (COUT / 2) + i;
        const float* sp = src + (size_t)c * HWS;
        float acc = 0.f;
        #pragma unroll
        for (int t = 0; t < 36; ++t)
            acc = fmaf(twgt[t], sp[toff[t]], acc);
        dst[(size_t)c * HWOUT] = acc;
    }
}

// ---------------------------------------------------------------------------
extern "C" void kernel_launch(void* const* d_in, const int* in_sizes, int n_in,
                              void* d_out, int out_size, void* d_ws, size_t ws_size,
                              hipStream_t stream)
{
    (void)in_sizes; (void)n_in; (void)out_size; (void)ws_size;
    const float* c1  = (const float*)d_in[0];   // [4,256,128,128]
    const float* c2  = (const float*)d_in[1];   // [4,512,64,64]
    const float* out = (const float*)d_in[2];   // [4,256,64,64]
    const float* w1  = (const float*)d_in[3];   // [32,256]
    const float* w2  = (const float*)d_in[4];   // [32,512]
    float* res = (float*)d_out;                 // [4,256,128,128]

    // workspace partition (~21.2 MB total; lives comfortably in 192 MB L2)
    float* qn  = (float*)d_ws;                  // [P][32]  normalized queries
    float* kfs = qn  + (size_t)PTOT  * CK;      // [P2][32] key field @64x64 (raw)
    float* kfn = kfs + (size_t)P2TOT * CK;      // [P][32]  upsampled+normalized keys
    float* att = kfn + (size_t)PTOT  * CK;      // [P][9]   softmax attention

    wmma_refine_kernel<256, HWOUT, true ><<<PTOT  / 128, 256, 0, stream>>>(c1, w1, qn);
    wmma_refine_kernel<512, HWS,   false><<<P2TOT / 128, 256, 0, stream>>>(c2, w2, kfs);
    upsample_norm_kernel <<<PTOT / 256, 256, 0, stream>>>(kfs, kfn);
    energy_softmax_kernel<<<PTOT / 256, 256, 0, stream>>>(qn, kfn, att);
    aggregate_kernel     <<<NIMG * HOUT, 256, 0, stream>>>(att, out, res);
}
